// AttentionSuper_13692355740281
// MI455X (gfx1250) — matrix operand order repaired and checked by reading, not compiled, not run on hardware
//
#include <hip/hip_runtime.h>

// ---------------------------------------------------------------------------
// CDNA5 (gfx1250) wave32 WMMA attention pipeline.
//   K1: qkv = x @ w_qkv^T            (f32 in, f16 out)   v_wmma_f32_16x16x32_f16
//   K2: flash attention per (b,h)    (f16 in, f16 out)   v_wmma_f32_16x16x32_f16
//   K3: out = attn @ w_proj^T + b    (f16 in, f32 out)   v_wmma_f32_16x16x32_f16
// f16-to-f16 tile staging uses global_load_async_to_lds_b128 (ASYNCcnt).
// ---------------------------------------------------------------------------

typedef __attribute__((ext_vector_type(16))) _Float16 v16h;
typedef __attribute__((ext_vector_type(8)))  _Float16 v8h;
typedef __attribute__((ext_vector_type(8)))  float    v8f;

union FragAB { v16h v; v8h h[2]; };

__device__ __forceinline__ v8f wmma_f16f32(v16h a, v16h b, v8f c) {
  // 8 args: (neg_a, A, neg_b, B, c_mod, C, reuse_a, reuse_b)
  return __builtin_amdgcn_wmma_f32_16x16x32_f16(false, a, false, b, (short)0, c,
                                                false, false);
}

// ---- staging helpers ------------------------------------------------------
// f32 source: load 8, convert to f16, store to LDS (plain VALU path).
__device__ __forceinline__ void stage8(const float* __restrict__ src,
                                       _Float16* dst) {
  const float4* s4 = reinterpret_cast<const float4*>(src);
  float4 f0 = s4[0], f1 = s4[1];
  dst[0] = (_Float16)f0.x; dst[1] = (_Float16)f0.y;
  dst[2] = (_Float16)f0.z; dst[3] = (_Float16)f0.w;
  dst[4] = (_Float16)f1.x; dst[5] = (_Float16)f1.y;
  dst[6] = (_Float16)f1.z; dst[7] = (_Float16)f1.w;
}
// f16 source: async DMA global -> LDS, 16 bytes, tracked by ASYNCcnt.
__device__ __forceinline__ void stage8(const _Float16* __restrict__ src,
                                       _Float16* dst) {
  asm volatile("global_load_async_to_lds_b128 %0, %1, off"
               :
               : "v"((unsigned int)(size_t)dst),
                 "v"((unsigned long long)(size_t)src)
               : "memory");
}

__device__ __forceinline__ void wait_async_then_barrier() {
  asm volatile("s_wait_asynccnt 0" ::: "memory");  // no-op if none issued
  __syncthreads();
}

// ---------------------------------------------------------------------------
// Generic GEMM: C[M,N] = X[M,K] * W[N,K]^T (+ bias), f32 accumulate via WMMA.
// Block: 256 threads = 8 waves; block tile 128(M) x 256(N); BK = 32.
// Wave grid 2x4, each wave owns a 64x64 tile = 4x4 WMMA fragments
// (16 wmma per K-step for 16 ds_load_b128 -> 1 DS load per WMMA).
// ---------------------------------------------------------------------------
template <typename TIn, typename TOut, bool HAS_BIAS>
__global__ __launch_bounds__(256) void gemm_wmma(
    const TIn* __restrict__ X, const float* __restrict__ W,
    TOut* __restrict__ C, const float* __restrict__ bias,
    int M, int N, int K) {
  constexpr int BM = 128, BN = 256, BK = 32, LDT = BK + 8;  // 40-half stride
  __shared__ alignas(16) _Float16 Xs[BM][LDT];
  __shared__ alignas(16) _Float16 Ws[BN][LDT];

  const int tid  = threadIdx.x;
  const int lane = tid & 31;
  const int wv   = tid >> 5;   // wave 0..7
  const int wm   = wv >> 2;    // 0..1  (64-row slice)
  const int wn   = wv & 3;     // 0..3  (64-col slice)
  const int ln   = lane & 15;
  const int lh   = lane >> 4;

  const int n0 = blockIdx.x * BN;
  const int m0 = blockIdx.y * BM;

  const v8f vzero = {};
  v8f acc[4][4];
#pragma unroll
  for (int ms = 0; ms < 4; ++ms)
#pragma unroll
    for (int ns = 0; ns < 4; ++ns) acc[ms][ns] = vzero;

  // cooperative load mapping
  const int xr = tid >> 1;        // 0..127
  const int xc = (tid & 1) * 16;  // 0,16
  const int wr = tid;             // 0..255 (one W row per thread)

  for (int k0 = 0; k0 < K; k0 += BK) {
    // ---- stage X tile: 16 elems/thread (async when TIn == f16) ----
    {
      const TIn* src = X + (size_t)(m0 + xr) * K + k0 + xc;
      stage8(src, &Xs[xr][xc]);
      stage8(src + 8, &Xs[xr][xc + 8]);
    }
    // ---- stage W tile: one full 32-elem row per thread (f32 -> f16) ----
    {
      const float* src = W + (size_t)(n0 + wr) * K + k0;
#pragma unroll
      for (int j = 0; j < 4; ++j) stage8(src + j * 8, &Ws[wr][j * 8]);
    }
    if (k0 + BK < K) {  // next-tile prefetch -> global_prefetch_b8
      __builtin_prefetch(X + (size_t)(m0 + xr) * K + k0 + BK + xc, 0, 1);
      __builtin_prefetch(W + (size_t)(n0 + wr) * K + k0 + BK, 0, 1);
    }
    wait_async_then_barrier();

    // ---- fragments ----
    FragAB a[4], b[4];
#pragma unroll
    for (int ms = 0; ms < 4; ++ms) {
      const int r = wm * 64 + ms * 16 + ln;  // A: lane holds row r
      // 16-bit A 16x32 layout: k = lh*8 + [0..7]  and  16 + lh*8 + [0..7]
      a[ms].h[0] = *reinterpret_cast<const v8h*>(&Xs[r][lh * 8]);
      a[ms].h[1] = *reinterpret_cast<const v8h*>(&Xs[r][16 + lh * 8]);
    }
#pragma unroll
    for (int ns = 0; ns < 4; ++ns) {
      const int r = wn * 64 + ns * 16 + ln;  // B: lane holds col r (row of W)
      // 16-bit B 32x16 layout: k = lh*8 + [0..7]  and  16 + lh*8 + [0..7]
      b[ns].h[0] = *reinterpret_cast<const v8h*>(&Ws[r][lh * 8]);
      b[ns].h[1] = *reinterpret_cast<const v8h*>(&Ws[r][16 + lh * 8]);
    }
#pragma unroll
    for (int ms = 0; ms < 4; ++ms)
#pragma unroll
      for (int ns = 0; ns < 4; ++ns)
        acc[ms][ns] = wmma_f16f32(a[ms].v, b[ns].v, acc[ms][ns]);

    __syncthreads();
  }

  // ---- epilogue: C layout — VGPR i: row i+lh*8, col = ln ----
#pragma unroll
  for (int ms = 0; ms < 4; ++ms)
#pragma unroll
    for (int ns = 0; ns < 4; ++ns)
#pragma unroll
      for (int i = 0; i < 8; ++i) {
        const int m = m0 + wm * 64 + ms * 16 + lh * 8 + i;
        const int n = n0 + wn * 64 + ns * 16 + ln;
        float v = acc[ms][ns][i];
        if constexpr (HAS_BIAS) v += bias[n];
        C[(size_t)m * N + n] = (TOut)v;
      }
  (void)bias;
}

// ---------------------------------------------------------------------------
// Flash attention. Grid: (L/64 row-blocks, B*H). Block: 128 threads = 4 waves,
// each wave owns 16 query rows. K-blocks of 64 keys staged in LDS (async
// copy); V staged transposed. Online softmax; P goes C-layout->LDS->A-layout.
// ---------------------------------------------------------------------------
__global__ __launch_bounds__(128) void flash_attn(
    const _Float16* __restrict__ qkv,  // [B*L, 2304] f16: [q|k|v] x H x 64
    const int* __restrict__ mask,      // [B, L]
    _Float16* __restrict__ out) {      // [B*L, 768]
  constexpr int L = 1024, H = 12, HD = 64, E3 = 2304, D = 768;
  const float SCALE = 0.125f;  // 64^-0.5

  __shared__ alignas(16) _Float16 Ks[64][72];      // keys  [k][d]
  __shared__ alignas(16) _Float16 Vt[64][72];      // V^T   [d][k]
  __shared__ alignas(16) _Float16 Pls[4][16][72];  // per-wave P staging

  const int tid  = threadIdx.x;
  const int lane = tid & 31;
  const int w    = tid >> 5;  // wave 0..3
  const int ln   = lane & 15;
  const int lh   = lane >> 4;

  const int rb = blockIdx.x;  // query row-block (64 rows)
  const int bh = blockIdx.y;
  const int b  = bh / H;
  const int h  = bh % H;

  // ---- Q fragments (held in registers for the whole kernel) ----
  const int qrow = rb * 64 + w * 16 + ln;
  const _Float16* qp = qkv + ((size_t)(b * L + qrow)) * E3 + h * HD;
  FragAB qa[2];
#pragma unroll
  for (int c = 0; c < 2; ++c) {
    qa[c].h[0] = *reinterpret_cast<const v8h*>(qp + c * 32 + lh * 8);
    qa[c].h[1] = *reinterpret_cast<const v8h*>(qp + c * 32 + 16 + lh * 8);
  }

  const v8f vzero = {};
  v8f o[4];
  o[0] = vzero; o[1] = vzero; o[2] = vzero; o[3] = vzero;
  float mprev[8], lsum[8];
#pragma unroll
  for (int i = 0; i < 8; ++i) { mprev[i] = -3.0e38f; lsum[i] = 0.f; }

  const int lr = tid >> 1;        // 0..63 (tile row)
  const int lc = (tid & 1) * 32;  // 0 / 32

  for (int kb = 0; kb < L / 64; ++kb) {
    // ---- stage K tile [64 keys][64 d] via async global->LDS DMA ----
    {
      const _Float16* src =
          qkv + ((size_t)(b * L + kb * 64 + lr)) * E3 + D + h * HD + lc;
#pragma unroll
      for (int j = 0; j < 4; ++j) stage8(src + j * 8, &Ks[lr][lc + j * 8]);
    }
    // ---- stage V transposed: Vt[d][k] (manual transpose) ----
    {
      const _Float16* src =
          qkv + ((size_t)(b * L + kb * 64 + lr)) * E3 + 2 * D + h * HD + lc;
      v8h tmp[4];
#pragma unroll
      for (int j4 = 0; j4 < 4; ++j4)
        tmp[j4] = *reinterpret_cast<const v8h*>(src + j4 * 8);
#pragma unroll
      for (int j = 0; j < 32; ++j) Vt[lc + j][lr] = tmp[j / 8][j % 8];
    }
    wait_async_then_barrier();

    // ---- S = Q * K^T  (16 x 64 per wave) ----
    v8f s[4];
    s[0] = vzero; s[1] = vzero; s[2] = vzero; s[3] = vzero;
#pragma unroll
    for (int nf = 0; nf < 4; ++nf) {
      const int kr = nf * 16 + ln;  // B-operand column = key row
#pragma unroll
      for (int c = 0; c < 2; ++c) {
        FragAB bk;
        bk.h[0] = *reinterpret_cast<const v8h*>(&Ks[kr][c * 32 + lh * 8]);
        bk.h[1] = *reinterpret_cast<const v8h*>(&Ks[kr][c * 32 + 16 + lh * 8]);
        s[nf] = wmma_f16f32(qa[c].v, bk.v, s[nf]);
      }
    }

    // ---- scale + mask ----
#pragma unroll
    for (int nf = 0; nf < 4; ++nf) {
      const int mkv = mask[(size_t)b * L + kb * 64 + nf * 16 + ln];
#pragma unroll
      for (int i = 0; i < 8; ++i) {
        const float v = s[nf][i] * SCALE;
        s[nf][i] = (mkv == 0) ? -3.0e38f : v;
      }
    }

    // ---- online softmax (rows live across 16-lane groups) ----
#pragma unroll
    for (int i = 0; i < 8; ++i) {
      float rm = fmaxf(fmaxf(s[0][i], s[1][i]), fmaxf(s[2][i], s[3][i]));
      rm = fmaxf(rm, __shfl_xor(rm, 1));
      rm = fmaxf(rm, __shfl_xor(rm, 2));
      rm = fmaxf(rm, __shfl_xor(rm, 4));
      rm = fmaxf(rm, __shfl_xor(rm, 8));
      const float mn    = fmaxf(mprev[i], rm);
      const float alpha = __expf(mprev[i] - mn);
      float rs = 0.f;
#pragma unroll
      for (int nf = 0; nf < 4; ++nf) {
        const float p = __expf(s[nf][i] - mn);
        s[nf][i] = p;
        rs += p;
      }
      rs += __shfl_xor(rs, 1);
      rs += __shfl_xor(rs, 2);
      rs += __shfl_xor(rs, 4);
      rs += __shfl_xor(rs, 8);
      lsum[i]  = lsum[i] * alpha + rs;
      mprev[i] = mn;
#pragma unroll
      for (int df = 0; df < 4; ++df) o[df][i] *= alpha;
    }

    // ---- P: C-layout f32 -> LDS f16 (row-major) -> A-layout frags ----
#pragma unroll
    for (int nf = 0; nf < 4; ++nf)
#pragma unroll
      for (int i = 0; i < 8; ++i)
        Pls[w][lh * 8 + i][nf * 16 + ln] = (_Float16)s[nf][i];
    asm volatile("s_wait_dscnt 0" ::: "memory");  // same-wave LDS RAW

    FragAB pa[2];
#pragma unroll
    for (int c = 0; c < 2; ++c) {
      pa[c].h[0] = *reinterpret_cast<const v8h*>(&Pls[w][ln][c * 32 + lh * 8]);
      pa[c].h[1] =
          *reinterpret_cast<const v8h*>(&Pls[w][ln][c * 32 + 16 + lh * 8]);
    }

    // ---- O += P * V  (keys are the K dim; B operand from Vt) ----
#pragma unroll
    for (int df = 0; df < 4; ++df) {
      const int dr = df * 16 + ln;  // output d index = B column
#pragma unroll
      for (int c = 0; c < 2; ++c) {
        FragAB bv;
        bv.h[0] = *reinterpret_cast<const v8h*>(&Vt[dr][c * 32 + lh * 8]);
        bv.h[1] = *reinterpret_cast<const v8h*>(&Vt[dr][c * 32 + 16 + lh * 8]);
        o[df] = wmma_f16f32(pa[c].v, bv.v, o[df]);
      }
    }
    __syncthreads();
  }

  // ---- normalize + store (f16, [B*L, 768] with head offset) ----
#pragma unroll
  for (int i = 0; i < 8; ++i) {
    const float inv = (lsum[i] > 0.f) ? (1.0f / lsum[i]) : 0.f;
    const int row = rb * 64 + w * 16 + lh * 8 + i;
#pragma unroll
    for (int df = 0; df < 4; ++df) {
      const int col = h * HD + df * 16 + ln;
      out[((size_t)(b * L + row)) * D + col] = (_Float16)(o[df][i] * inv);
    }
  }
}

// ---------------------------------------------------------------------------
// Host-side launcher. Inputs: x, attn_mask, w_qkv, w_proj, b_proj.
// Workspace: qkv f16 (8192x2304) then attn-out f16 (8192x768).
// ---------------------------------------------------------------------------
extern "C" void kernel_launch(void* const* d_in, const int* in_sizes, int n_in,
                              void* d_out, int out_size, void* d_ws,
                              size_t ws_size, hipStream_t stream) {
  (void)in_sizes; (void)n_in; (void)out_size; (void)ws_size;
  const float* x      = (const float*)d_in[0];
  const int*   amask  = (const int*)d_in[1];
  const float* w_qkv  = (const float*)d_in[2];
  const float* w_proj = (const float*)d_in[3];
  const float* b_proj = (const float*)d_in[4];
  float*       outp   = (float*)d_out;

  constexpr int Brows = 8 * 1024;  // 8192
  constexpr int D = 768, E3 = 2304;

  _Float16* qkv = (_Float16*)d_ws;
  _Float16* att =
      (_Float16*)((char*)d_ws + (size_t)Brows * E3 * sizeof(_Float16));

  // K1: qkv = x @ w_qkv^T   (M=8192, N=2304, K=768)
  gemm_wmma<float, _Float16, false>
      <<<dim3(E3 / 256, Brows / 128), dim3(256), 0, stream>>>(
          x, w_qkv, qkv, nullptr, Brows, E3, D);

  // K2: flash attention (16 row-blocks x 96 (b,h) pairs)
  flash_attn<<<dim3(16, 96), dim3(128), 0, stream>>>(qkv, amask, att);

  // K3: out = att @ w_proj^T + b_proj   (M=8192, N=768, K=768)
  gemm_wmma<_Float16, float, true>
      <<<dim3(D / 256, Brows / 128), dim3(256), 0, stream>>>(
          att, w_proj, outp, b_proj, Brows, D, D);
}
